// Attention_1460288880914
// MI455X (gfx1250) — matrix operand order repaired and checked by reading, not compiled, not run on hardware
//
#include <hip/hip_runtime.h>
#include <hip/hip_bf16.h>

// ---------------- CDNA5 WMMA types ----------------
typedef __attribute__((ext_vector_type(16))) __bf16 v16bf;
typedef __attribute__((ext_vector_type(8)))  __bf16 v8bf;
typedef __attribute__((ext_vector_type(4)))  __bf16 v4bf;
typedef __attribute__((ext_vector_type(8)))  float  v8f;

__device__ __forceinline__ v8f wmma_bf16(v16bf a, v16bf b, v8f c) {
  // D = A(16x32) * B(32x16) + C, f32 accumulate
  return __builtin_amdgcn_wmma_f32_16x16x32_bf16(false, a, false, b,
                                                 (short)0, c, false, false);
}

union U16 { v16bf v; v8bf h[2]; __bf16 e[16]; };

#define S_LEN 2048
#define DIM   4096
#define NH    32
#define NKV   8
#define HD    128
#define NEGV  (-1000000000.0f)

// CDNA5 async memory->LDS copy (VGLOBAL GVS mode, tracked by ASYNCcnt).
__device__ __forceinline__ void async_lds_b128(unsigned lds_off, unsigned goff,
                                               const void* base) {
  asm volatile("global_load_async_to_lds_b128 %0, %1, %2"
               :
               : "v"(lds_off), "v"(goff), "s"((unsigned long long)base)
               : "memory");
}
#define WAIT_ASYNC_4() asm volatile("s_wait_asynccnt 0x4" ::: "memory")
#define WAIT_ASYNC_0() asm volatile("s_wait_asynccnt 0x0" ::: "memory")

// ---------------- f32 -> bf16 cast ----------------
__global__ void cast_f32_bf16(const float* __restrict__ in,
                              __bf16* __restrict__ out, int n4) {
  int i = blockIdx.x * 256 + threadIdx.x;
  if (i < n4) {
    float4 v = ((const float4*)in)[i];
    v4bf o = { (__bf16)v.x, (__bf16)v.y, (__bf16)v.z, (__bf16)v.w };
    ((v4bf*)out)[i] = o;
  }
}

// ---------------- GEMM: C[M,N] = A[M,K] * W[N,K]^T -----------------
// 128x128 block tile, BK=32, 8 waves: 4 (M) x 2 (N), wave tile 32x64.
// Double-buffered LDS filled by global_load_async_to_lds_b128; the pipeline
// loop is unrolled by 2 so the buffer index is a compile-time constant and
// the steady state has no branches: prefetch -> wait_asynccnt 4 -> barrier
// -> frags/WMMA -> barrier. K/32 must be even and >= 4 (K is 4096 here).
#define LDT 40   // padded LDS row stride (bf16 elems), 16B-aligned
__global__ __launch_bounds__(256)
void gemm_bf16_nt(const __bf16* __restrict__ A, const __bf16* __restrict__ W,
                  float* __restrict__ C, int M, int N, int K) {
  __shared__ __bf16 sA[2][128 * LDT];
  __shared__ __bf16 sB[2][128 * LDT];

  const int tid  = threadIdx.x;
  const int lane = tid & 31;
  const int wave = tid >> 5;
  const int wm   = wave & 3;   // 4 waves along M
  const int wn   = wave >> 2;  // 2 waves along N
  const int bm   = blockIdx.y * 128;
  const int bn   = blockIdx.x * 128;
  const int r    = lane & 15;
  const int hf   = lane >> 4;

  v8f acc[2][4];
#pragma unroll
  for (int i = 0; i < 2; i++)
#pragma unroll
    for (int j = 0; j < 4; j++) acc[i][j] = (v8f){0,0,0,0,0,0,0,0};

  const int ldrow = tid >> 1;           // 0..127
  const int ldcol = (tid & 1) * 16;     // 0 or 16

  // byte offsets for this thread's slice of each tile
  const unsigned gaA = (unsigned)(((bm + ldrow) * K + ldcol) * 2);
  const unsigned gaB = (unsigned)(((bn + ldrow) * K + ldcol) * 2);
  unsigned lA[2], lB[2];
  lA[0] = (unsigned)(unsigned long long)(const void*)&sA[0][ldrow * LDT + ldcol];
  lA[1] = (unsigned)(unsigned long long)(const void*)&sA[1][ldrow * LDT + ldcol];
  lB[0] = (unsigned)(unsigned long long)(const void*)&sB[0][ldrow * LDT + ldcol];
  lB[1] = (unsigned)(unsigned long long)(const void*)&sB[1][ldrow * LDT + ldcol];

  // prefetch tile `it` into LDS buffer `buf` (4 async ops per thread)
  auto prefetch = [&](int buf, int it) {
    const unsigned kb = (unsigned)(it * 64);       // k0*2 bytes
    async_lds_b128(lA[buf],      gaA + kb,      A);
    async_lds_b128(lA[buf] + 16, gaA + kb + 16, A);
    async_lds_b128(lB[buf],      gaB + kb,      W);
    async_lds_b128(lB[buf] + 16, gaB + kb + 16, W);
  };

  // consume tile in LDS buffer `buf` (compile-time constant after inlining)
  auto process = [&](int buf) {
    v16bf af[2];
#pragma unroll
    for (int i = 0; i < 2; i++) {
      // A frag: lane<16 -> K 0..7 & 16..23 ; lane>=16 -> K 8..15 & 24..31
      const __bf16* p = &sA[buf][(wm * 32 + i * 16 + r) * LDT + hf * 8];
      U16 u; u.h[0] = *(const v8bf*)p; u.h[1] = *(const v8bf*)(p + 16);
      af[i] = u.v;
    }
    v16bf bf[4];
#pragma unroll
    for (int j = 0; j < 4; j++) {
      // B frag: lane<16 -> K 0..7 & 16..23 ; lane>=16 -> K 8..15 & 24..31
      const __bf16* p = &sB[buf][(wn * 64 + j * 16 + r) * LDT + hf * 8];
      U16 u; u.h[0] = *(const v8bf*)p; u.h[1] = *(const v8bf*)(p + 16);
      bf[j] = u.v;
    }
#pragma unroll
    for (int i = 0; i < 2; i++)
#pragma unroll
      for (int j = 0; j < 4; j++)
        acc[i][j] = wmma_bf16(af[i], bf[j], acc[i][j]);
  };

  const int nk = K >> 5;   // even, >= 4 for all launches here
  prefetch(0, 0);

  // steady state, unrolled x2: tiles it (buf0) and it+1 (buf1)
  for (int it = 0; it + 2 < nk; it += 2) {
    prefetch(1, it + 1);
    WAIT_ASYNC_4();          // tile it landed (in-order async completion)
    __syncthreads();
    process(0);
    __syncthreads();

    prefetch(0, it + 2);
    WAIT_ASYNC_4();          // tile it+1 landed
    __syncthreads();
    process(1);
    __syncthreads();
  }
  // peeled tail: tiles nk-2 (buf0) and nk-1 (buf1)
  prefetch(1, nk - 1);
  WAIT_ASYNC_4();
  __syncthreads();
  process(0);
  __syncthreads();

  WAIT_ASYNC_0();
  __syncthreads();
  process(1);

  // epilogue: base + 32-bit offsets -> SADDR-form stores
#pragma unroll
  for (int i = 0; i < 2; i++) {
#pragma unroll
    for (int j = 0; j < 4; j++) {
      unsigned rowb = (unsigned)(bm + wm * 32 + i * 16 + hf * 8);
      unsigned col  = (unsigned)(bn + wn * 64 + j * 16 + r);
#pragma unroll
      for (int e = 0; e < 8; e++)
        C[(rowb + e) * (unsigned)N + col] = acc[i][j][e];
    }
  }
}

// ---------------- RoPE + cache write + attention-layout buffers ----------
__global__ __launch_bounds__(256)
void rope_cache_kernel(const float* __restrict__ xq, const float* __restrict__ xk,
                       const float* __restrict__ xv,
                       const float* __restrict__ fcos, const float* __restrict__ fsin,
                       const long long* __restrict__ idx,
                       __bf16* __restrict__ qb,   // [S][NH][HD] roped
                       __bf16* __restrict__ kb,   // [S][NKV][HD] roped
                       __bf16* __restrict__ vT,   // [NKV][HD][S]
                       float* __restrict__ cache_k, float* __restrict__ cache_v) {
  const int s = blockIdx.x;
  const int t = threadIdx.x;
  const unsigned pos = (unsigned)idx[s];

  // Q: 2048 pairs per row
  for (int p = t; p < 2048; p += 256) {
    int d2 = p & 63;
    float c  = fcos[s * 64 + d2];
    float sn = fsin[s * 64 + d2];
    float xr = xq[(unsigned)(s * DIM + 2 * p)];
    float xi = xq[(unsigned)(s * DIM + 2 * p + 1)];
    qb[(unsigned)(s * DIM + 2 * p)]     = (__bf16)(xr * c - xi * sn);
    qb[(unsigned)(s * DIM + 2 * p + 1)] = (__bf16)(xr * sn + xi * c);
  }
  // K: 512 pairs per row (rope + f32 cache + bf16 buffer)
  for (int p = t; p < 512; p += 256) {
    int d2 = p & 63;
    float c  = fcos[s * 64 + d2];
    float sn = fsin[s * 64 + d2];
    float xr = xk[(unsigned)(s * 1024 + 2 * p)];
    float xi = xk[(unsigned)(s * 1024 + 2 * p + 1)];
    float outr = xr * c - xi * sn;
    float outi = xr * sn + xi * c;
    kb[(unsigned)(s * 1024 + 2 * p)]     = (__bf16)outr;
    kb[(unsigned)(s * 1024 + 2 * p + 1)] = (__bf16)outi;
    cache_k[pos * 1024u + 2 * p]     = outr;
    cache_k[pos * 1024u + 2 * p + 1] = outi;
  }
  // V: f32 cache copy + transposed bf16 buffer
  for (int e = t; e < 1024; e += 256) {
    float val = xv[(unsigned)(s * 1024 + e)];
    cache_v[pos * 1024u + e] = val;
    int kvh = e >> 7, d = e & 127;
    vT[(unsigned)((kvh * 128 + d) * S_LEN + s)] = (__bf16)val;
  }
}

// ---------------- Flash attention (causal, GQA 4:1) ----------------
// One wave = 16 queries of one head; Sᵀ = K·Qᵀ so softmax is lane-local;
// Sᵀ C-layout == P A-layout over 32 keys after bf16 packing (no shuffles).
__global__ __launch_bounds__(256)
void attn_kernel(const __bf16* __restrict__ qb, const __bf16* __restrict__ kb,
                 const __bf16* __restrict__ vT, __bf16* __restrict__ ob) {
  const int h    = blockIdx.x;             // head
  const int kvh  = h >> 2;                 // N_REP = 4
  const int wave = threadIdx.x >> 5;
  const int lane = threadIdx.x & 31;
  const int r    = lane & 15;
  const int hf   = lane >> 4;
  const int q0   = blockIdx.y * 128 + wave * 16;
  const float scale = 0.08838834764831845f; // 1/sqrt(128)

  // preload Q B-fragments (B cols = queries, table order over HD per lane)
  v16bf qf[4];
  const unsigned qoff = (unsigned)((q0 + r) * DIM + h * HD);
#pragma unroll
  for (int kk = 0; kk < 4; kk++) {
    const __bf16* p = qb + qoff + kk * 32 + hf * 8;
    U16 u; u.h[0] = *(const v8bf*)p; u.h[1] = *(const v8bf*)(p + 16);
    qf[kk] = u.v;
  }

  v8f o[8];
#pragma unroll
  for (int n = 0; n < 8; n++) o[n] = (v8f){0,0,0,0,0,0,0,0};
  float m = -1e30f, l = 0.0f;
  const int qcol = q0 + r;
  const int kend = q0 + 16;

  for (int k0 = 0; k0 < kend; k0 += 32) {
    v8f st0 = (v8f){0,0,0,0,0,0,0,0};
    v8f st1 = (v8f){0,0,0,0,0,0,0,0};
#pragma unroll
    for (int kk = 0; kk < 4; kk++) {
      // A = K rows: lane<16 -> dims kk*32+{0..7,16..23}; hi lanes shifted by 8
      const __bf16* k0p = kb + (unsigned)((k0 + r) * 1024 + kvh * HD + kk * 32 + hf * 8);
      U16 ua; ua.h[0] = *(const v8bf*)k0p; ua.h[1] = *(const v8bf*)(k0p + 16);
      const __bf16* k1p = kb + (unsigned)((k0 + 16 + r) * 1024 + kvh * HD + kk * 32 + hf * 8);
      U16 ub; ub.h[0] = *(const v8bf*)k1p; ub.h[1] = *(const v8bf*)(k1p + 16);
      st0 = wmma_bf16(ua.v, qf[kk], st0);
      st1 = wmma_bf16(ub.v, qf[kk], st1);
    }

    // scale + causal mask; lane holds 16 score values for query `qcol`
    float sc[16];
#pragma unroll
    for (int e = 0; e < 8; e++) {
      int key0 = k0 + e + hf * 8;
      int key1 = key0 + 16;
      sc[e]     = st0[e] * scale + ((key0 > qcol) ? NEGV : 0.0f);
      sc[8 + e] = st1[e] * scale + ((key1 > qcol) ? NEGV : 0.0f);
    }
    // online softmax (combine lane pair via xor-16)
    float bm = sc[0];
#pragma unroll
    for (int i = 1; i < 16; i++) bm = fmaxf(bm, sc[i]);
    bm = fmaxf(bm, __shfl_xor(bm, 16, 32));
    float mnew  = fmaxf(m, bm);
    float alpha = __expf(m - mnew);
    m = mnew;
    float ls = 0.0f;
#pragma unroll
    for (int i = 0; i < 16; i++) { sc[i] = __expf(sc[i] - mnew); ls += sc[i]; }
    l = l * alpha + ls;

    // rescale O rows (row = query j + 8*hf; alpha lives in lane j (+16))
    float aj[8];
#pragma unroll
    for (int j = 0; j < 8; j++) aj[j] = __shfl(alpha, j + hf * 8, 32);
#pragma unroll
    for (int n = 0; n < 8; n++)
#pragma unroll
      for (int j = 0; j < 8; j++) o[n][j] *= aj[j];

    // pack P: Sᵀ C-layout reinterprets directly as P A-layout over 32 keys
    U16 pu;
#pragma unroll
    for (int i = 0; i < 16; i++) pu.e[i] = (__bf16)sc[i];

    // PV: B = V (cols = head dims) from Vᵀ, table order over keys per lane
#pragma unroll
    for (int n = 0; n < 8; n++) {
      const __bf16* vp = vT + (unsigned)((kvh * 128 + n * 16 + r) * S_LEN + k0 + hf * 8);
      U16 uv; uv.h[0] = *(const v8bf*)vp; uv.h[1] = *(const v8bf*)(vp + 16);
      o[n] = wmma_bf16(pu.v, uv.v, o[n]);
    }
  }

  // finalize: combine row sums across lane pair, normalize, store bf16
  float lt   = l + __shfl_xor(l, 16, 32);
  float linv = 1.0f / lt;
  float lj[8];
#pragma unroll
  for (int j = 0; j < 8; j++) lj[j] = __shfl(linv, j + hf * 8, 32);
#pragma unroll
  for (int n = 0; n < 8; n++) {
#pragma unroll
    for (int j = 0; j < 8; j++) {
      unsigned row = (unsigned)(q0 + j + hf * 8);
      unsigned col = (unsigned)(n * 16 + r);
      ob[row * (unsigned)DIM + h * HD + col] = (__bf16)(o[n][j] * lj[j]);
    }
  }
}

// ---------------- host-side launch ----------------
extern "C" void kernel_launch(void* const* d_in, const int* in_sizes, int n_in,
                              void* d_out, int out_size, void* d_ws, size_t ws_size,
                              hipStream_t stream) {
  const float*     x    = (const float*)d_in[0];
  const float*     fcos = (const float*)d_in[1];
  const float*     fsin = (const float*)d_in[2];
  const long long* idx  = (const long long*)d_in[4];
  const float*     wq   = (const float*)d_in[7];
  const float*     wk   = (const float*)d_in[8];
  const float*     wv   = (const float*)d_in[9];
  const float*     wo   = (const float*)d_in[10];

  float* out     = (float*)d_out;
  float* cache_k = out + (size_t)S_LEN * DIM;
  float* cache_v = cache_k + (size_t)S_LEN * NKV * HD;

  // workspace bump allocator (total ~193 MB)
  char* wsp = (char*)d_ws;
  auto alloc = [&](size_t bytes) {
    char* p = wsp; wsp += (bytes + 255) & ~(size_t)255; return (void*)p;
  };
  const size_t NXD = (size_t)S_LEN * DIM;      // 2048*4096
  const size_t NKD = (size_t)S_LEN * 1024;     // 2048*1024
  __bf16* xb    = (__bf16*)alloc(NXD * 2);
  __bf16* wqb   = (__bf16*)alloc((size_t)DIM * DIM * 2);
  __bf16* wkb   = (__bf16*)alloc((size_t)1024 * DIM * 2);
  __bf16* wvb   = (__bf16*)alloc((size_t)1024 * DIM * 2);
  __bf16* wob   = (__bf16*)alloc((size_t)DIM * DIM * 2);
  float*  xqf   = (float*) alloc(NXD * 4);
  float*  xkf   = (float*) alloc(NKD * 4);
  float*  xvf   = (float*) alloc(NKD * 4);
  __bf16* qbuf  = (__bf16*)alloc(NXD * 2);
  __bf16* kbuf  = (__bf16*)alloc(NKD * 2);
  __bf16* vTbuf = (__bf16*)alloc(NKD * 2);
  __bf16* abuf  = (__bf16*)alloc(NXD * 2);

  // 1) casts
  auto cgrid = [](size_t n) { return (unsigned)((n / 4 + 255) / 256); };
  cast_f32_bf16<<<cgrid(NXD), 256, 0, stream>>>(x,  xb,  (int)(NXD / 4));
  cast_f32_bf16<<<cgrid((size_t)DIM * DIM), 256, 0, stream>>>(wq, wqb, DIM * DIM / 4);
  cast_f32_bf16<<<cgrid((size_t)1024 * DIM), 256, 0, stream>>>(wk, wkb, 1024 * DIM / 4);
  cast_f32_bf16<<<cgrid((size_t)1024 * DIM), 256, 0, stream>>>(wv, wvb, 1024 * DIM / 4);
  cast_f32_bf16<<<cgrid((size_t)DIM * DIM), 256, 0, stream>>>(wo, wob, DIM * DIM / 4);

  // 2) QKV projections
  gemm_bf16_nt<<<dim3(DIM / 128, S_LEN / 128), 256, 0, stream>>>(xb, wqb, xqf, S_LEN, DIM, DIM);
  gemm_bf16_nt<<<dim3(1024 / 128, S_LEN / 128), 256, 0, stream>>>(xb, wkb, xkf, S_LEN, 1024, DIM);
  gemm_bf16_nt<<<dim3(1024 / 128, S_LEN / 128), 256, 0, stream>>>(xb, wvb, xvf, S_LEN, 1024, DIM);

  // 3) RoPE + caches + attention buffers
  rope_cache_kernel<<<S_LEN, 256, 0, stream>>>(xqf, xkf, xvf, fcos, fsin, idx,
                                               qbuf, kbuf, vTbuf, cache_k, cache_v);

  // 4) causal flash attention
  attn_kernel<<<dim3(NH, S_LEN / 128), 256, 0, stream>>>(qbuf, kbuf, vTbuf, abuf);

  // 5) output projection -> d_out
  gemm_bf16_nt<<<dim3(DIM / 128, S_LEN / 128), 256, 0, stream>>>(abuf, wob, out, S_LEN, DIM, DIM);
}
